// MultiheadAttention_78091095376177
// MI455X (gfx1250) — matrix operand order repaired and checked by reading, not compiled, not run on hardware
//
#include <hip/hip_runtime.h>

// ---------------------------------------------------------------------------
// MI455X (gfx1250) multi-head attention, bf16 WMMA path.
// B=4, S=2048, D=1024, H=16, HD=64
// ---------------------------------------------------------------------------

#define B_  4
#define S_  2048
#define D_  1024
#define H_  16
#define HD_ 64
#define BS_ (B_ * S_)

typedef __bf16 bf16_t;
typedef __attribute__((ext_vector_type(16))) __bf16 v16bf;
typedef __attribute__((ext_vector_type(8)))  float  v8f;

// ---- fragment helpers ------------------------------------------------------
// A-operand (16x32 bf16): lane r=lane&15 is row M, half=lane>>4 selects
// K-runs [half*8 .. +7] and [16+half*8 .. +7]  (two contiguous 16B loads).
// B-operand (32x16 bf16): lane r is column N, elements in the K order of the paired A
// (one contiguous 32B load).

__device__ __forceinline__ v16bf ld_frag32(const bf16_t* p) {
  union { v16bf v; uint4 q[2]; } u;
  u.q[0] = ((const uint4*)p)[0];
  u.q[1] = ((const uint4*)p)[1];
  return u.v;
}

__device__ __forceinline__ v16bf ld_frag2x16(const bf16_t* p0, const bf16_t* p1) {
  union { v16bf v; uint4 q[2]; } u;
  u.q[0] = *(const uint4*)p0;
  u.q[1] = *(const uint4*)p1;
  return u.v;
}

__device__ __forceinline__ v8f wmma_bf16(v16bf a, v16bf b, v8f c) {
  return __builtin_amdgcn_wmma_f32_16x16x32_bf16(
      /*neg_a=*/false, a, /*neg_b=*/false, b,
      /*c_mod=*/(short)0, c, /*reuse_a=*/false, /*reuse_b=*/false);
}

// ---------------------------------------------------------------------------
// Kernel 1: pack fp32 weights [n][k] into bf16 B-fragment layout.
// Per (kt,nt) 32x16 tile: 32 lanes x 16 bf16 contiguous; lane L packs
// W[nt*16 + (L&15)][kt*32 + (L>>4)*8 + (e&7) + 16*(e>>3)], e = 0..15 (two 8-runs in W row).
// ---------------------------------------------------------------------------
__global__ __launch_bounds__(256)
void pack_weights(const float* __restrict__ W0, const float* __restrict__ W1,
                  const float* __restrict__ W2, const float* __restrict__ W3,
                  bf16_t* __restrict__ out) {
  const int task = blockIdx.x * 8 + (threadIdx.x >> 5);
  const int lane = threadIdx.x & 31;
  const int mat  = task >> 11;        // 2048 tiles per matrix
  const int rem  = task & 2047;
  const int kt   = rem >> 6;
  const int nt   = rem & 63;

  const float* W = (mat == 0) ? W0 : (mat == 1) ? W1 : (mat == 2) ? W2 : W3;
  const int n = nt * 16 + (lane & 15);
  const int k = kt * 32 + (lane >> 4) * 8;

  const float* src = W + (size_t)n * D_ + k;
  bf16_t* dst = out + (size_t)mat * D_ * D_ + ((size_t)rem * 32 + lane) * 16;
#pragma unroll
  for (int e = 0; e < 16; ++e) dst[e] = (__bf16)src[(e & 7) + ((e >> 3) << 4)];
}

// ---------------------------------------------------------------------------
// Kernel 2: Y = X(fp32) @ W^T + b  ->  bf16, scattered per MODE.
//   MODE 0: [B,H,S,HD]   (Q, K)
//   MODE 1: [B,H,HD,S]   (V transposed, so PV B-operand is contiguous)
// Workgroup: 8 waves, 128(M) x 128(N) tile; wave = 32x64 (2x4 WMMA accs).
// K unrolled by 2 -> 16 WMMAs per loop body.
// ---------------------------------------------------------------------------
template <int MODE>
__global__ __launch_bounds__(256)
void proj_gemm(const float* __restrict__ X, const bf16_t* __restrict__ Wp,
               const float* __restrict__ bias, bf16_t* __restrict__ Y) {
  const int lane = threadIdx.x & 31;
  const int wave = threadIdx.x >> 5;
  const int r    = lane & 15;
  const int half = lane >> 4;
  const int m0   = blockIdx.x * 128 + (wave & 3) * 32;
  const int n0   = blockIdx.y * 128 + (wave >> 2) * 64;

  v8f acc[2][4] = {};

  for (int k0 = 0; k0 < D_; k0 += 64) {
#pragma unroll
    for (int kk = 0; kk < 2; ++kk) {
      const int k = k0 + kk * 32;
      v16bf a[2], bfr[4];
#pragma unroll
      for (int mt = 0; mt < 2; ++mt) {
        const float* ap = X + (size_t)(m0 + mt * 16 + r) * D_ + k;
        __builtin_prefetch(ap + 64, 0, 1);       // next K-tile of A
        v16bf t;
#pragma unroll
        for (int e = 0; e < 8; ++e) t[e]     = (__bf16)ap[half * 8 + e];
#pragma unroll
        for (int e = 0; e < 8; ++e) t[8 + e] = (__bf16)ap[16 + half * 8 + e];
        a[mt] = t;
      }
      const int kt = k >> 5;
#pragma unroll
      for (int nt = 0; nt < 4; ++nt) {
        const int ntile = (n0 >> 4) + nt;
        const bf16_t* bp = Wp + ((size_t)(kt * (D_ / 16) + ntile) * 32 + lane) * 16;
        __builtin_prefetch(bp + 2 * (D_ / 16) * 32 * 16, 0, 1);  // next K-tile of W
        bfr[nt] = ld_frag32(bp);
      }
#pragma unroll
      for (int mt = 0; mt < 2; ++mt)
#pragma unroll
        for (int nt = 0; nt < 4; ++nt)
          acc[mt][nt] = wmma_bf16(a[mt], bfr[nt], acc[mt][nt]);
    }
  }

#pragma unroll
  for (int mt = 0; mt < 2; ++mt) {
#pragma unroll
    for (int nt = 0; nt < 4; ++nt) {
      const int gn = n0 + nt * 16 + r;
      const float bv = bias[gn];
      const int h = gn >> 6, hd = gn & 63;
#pragma unroll
      for (int i = 0; i < 8; ++i) {
        const int gm = m0 + mt * 16 + i + half * 8;   // C layout: M = i + 8*half
        const int b = gm >> 11, s = gm & (S_ - 1);
        const float val = acc[mt][nt][i] + bv;
        size_t idx;
        if (MODE == 0) idx = ((size_t)(b * H_ + h) * S_ + s) * HD_ + hd;
        else           idx = ((size_t)(b * H_ + h) * HD_ + hd) * S_ + s;
        Y[idx] = (__bf16)val;
      }
    }
  }
}

// ---------------------------------------------------------------------------
// Kernel 3: causal flash attention.
// Grid: (S/128, B*H). 8 waves/block; each wave owns 16 query rows,
// keeps Q (16x64 bf16, pre-scaled by 1/sqrt(HD) -- exact: 2^-3) + O (16x64
// f32) in registers, streams KV in 32-column chunks with online softmax.
// P re-layout bounces via per-wave LDS scratch.
// ---------------------------------------------------------------------------
__global__ __launch_bounds__(256)
void flash_attn(const bf16_t* __restrict__ Qb, const bf16_t* __restrict__ Kb,
                const bf16_t* __restrict__ Vt, bf16_t* __restrict__ Ob) {
  __shared__ __bf16 ldsP[8 * 16 * 32];

  const int lane = threadIdx.x & 31;
  const int wave = threadIdx.x >> 5;
  const int r    = lane & 15;
  const int half = lane >> 4;
  const int bh   = blockIdx.y;                 // b*H + h
  const int b    = bh >> 4, h = bh & 15;
  const int q0   = blockIdx.x * 128 + wave * 16;

  // Q A-fragments for HD chunks [0,32) and [32,64), pre-scaled by 0.125
  const bf16_t* Qrow = Qb + ((size_t)bh * S_ + q0 + r) * HD_;
  v16bf qf[2];
#pragma unroll
  for (int c = 0; c < 2; ++c) {
    v16bf t = ld_frag2x16(Qrow + c * 32 + half * 16, Qrow + c * 32 + half * 16 + 8);
#pragma unroll
    for (int e = 0; e < 16; ++e) t[e] = (__bf16)((float)t[e] * 0.125f);
    qf[c] = t;
  }

  v8f o[4] = {};
  float m8[8], l8[8];
#pragma unroll
  for (int i = 0; i < 8; ++i) { m8[i] = -1e30f; l8[i] = 0.0f; }

  __bf16* pw = ldsP + wave * (16 * 32);
  const int jend = q0 + 16;                    // causal: cols <= q0+15

  for (int j0 = 0; j0 < jend; j0 += 32) {
    const bf16_t* krow = Kb + ((size_t)bh * S_ + j0 + r) * HD_;
    // prefetch next KV chunk (speculative; OOB tails are dropped by HW)
    __builtin_prefetch(krow + 32 * HD_, 0, 1);
    __builtin_prefetch(Vt + ((size_t)bh * HD_ + r) * S_ + j0 + 32, 0, 1);

    // logits S = Q @ K^T over two 16-wide column tiles
    v8f s0 = {}, s1 = {};
#pragma unroll
    for (int c = 0; c < 2; ++c) {
      s0 = wmma_bf16(qf[c], ld_frag32(krow + c * 32 + half * 16), s0);
      s1 = wmma_bf16(qf[c], ld_frag32(krow + 16 * HD_ + c * 32 + half * 16), s1);
    }

    float p0[8], p1[8], alpha[8];
#pragma unroll
    for (int i = 0; i < 8; ++i) {
      const int qr = q0 + i + half * 8;        // this lane-half's row
      float v0 = s0[i];
      float v1 = s1[i];
      if (j0 + r      > qr) v0 = -1e9f;        // causal mask
      if (j0 + 16 + r > qr) v1 = -1e9f;
      float t = fmaxf(v0, v1);
#pragma unroll
      for (int k = 1; k < 16; k <<= 1) t = fmaxf(t, __shfl_xor(t, k, 32));
      const float mn = fmaxf(m8[i], t);
      alpha[i] = __expf(m8[i] - mn);
      p0[i] = __expf(v0 - mn);
      p1[i] = __expf(v1 - mn);
      float rs = p0[i] + p1[i];
#pragma unroll
      for (int k = 1; k < 16; k <<= 1) rs += __shfl_xor(rs, k, 32);
      l8[i] = l8[i] * alpha[i] + rs;
      m8[i] = mn;
    }
#pragma unroll
    for (int t = 0; t < 4; ++t)
#pragma unroll
      for (int i = 0; i < 8; ++i) o[t][i] *= alpha[i];

    // C-layout -> A-fragment re-layout through per-wave LDS scratch
#pragma unroll
    for (int i = 0; i < 8; ++i) {
      const int row = i + half * 8;
      pw[row * 32 + r]      = (__bf16)p0[i];
      pw[row * 32 + 16 + r] = (__bf16)p1[i];
    }
    const v16bf pf = ld_frag2x16(pw + r * 32 + half * 16,
                                 pw + r * 32 + half * 16 + 8);

    // O += P @ V   (V stored [B,H,HD,S] so B-operand loads are contiguous)
#pragma unroll
    for (int t = 0; t < 4; ++t) {
      const bf16_t* vp = Vt + ((size_t)bh * HD_ + t * 16 + r) * S_ + j0 + half * 16;
      o[t] = wmma_bf16(pf, ld_frag32(vp), o[t]);
    }
  }

  // normalize and store attention output as bf16 [B,S,D] (d = h*64+hd)
#pragma unroll
  for (int t = 0; t < 4; ++t) {
#pragma unroll
    for (int i = 0; i < 8; ++i) {
      const int s = q0 + i + half * 8;
      const float val = o[t][i] / l8[i];
      Ob[((size_t)(b * S_ + s)) * D_ + h * 64 + t * 16 + r] = (__bf16)val;
    }
  }
}

// ---------------------------------------------------------------------------
// Kernel 4: out = O(bf16) @ Wo^T + bo  -> fp32.  Same 128x128 tiling.
// ---------------------------------------------------------------------------
__global__ __launch_bounds__(256)
void out_gemm(const bf16_t* __restrict__ A, const bf16_t* __restrict__ Wp,
              const float* __restrict__ bias, float* __restrict__ Y) {
  const int lane = threadIdx.x & 31;
  const int wave = threadIdx.x >> 5;
  const int r    = lane & 15;
  const int half = lane >> 4;
  const int m0   = blockIdx.x * 128 + (wave & 3) * 32;
  const int n0   = blockIdx.y * 128 + (wave >> 2) * 64;

  v8f acc[2][4] = {};

  for (int k0 = 0; k0 < D_; k0 += 64) {
#pragma unroll
    for (int kk = 0; kk < 2; ++kk) {
      const int k = k0 + kk * 32;
      v16bf a[2], bfr[4];
#pragma unroll
      for (int mt = 0; mt < 2; ++mt) {
        const bf16_t* ap = A + (size_t)(m0 + mt * 16 + r) * D_ + k;
        __builtin_prefetch(ap + 64, 0, 1);
        a[mt] = ld_frag2x16(ap + half * 8, ap + 16 + half * 8);
      }
      const int kt = k >> 5;
#pragma unroll
      for (int nt = 0; nt < 4; ++nt) {
        const int ntile = (n0 >> 4) + nt;
        const bf16_t* bp = Wp + ((size_t)(kt * (D_ / 16) + ntile) * 32 + lane) * 16;
        __builtin_prefetch(bp + 2 * (D_ / 16) * 32 * 16, 0, 1);
        bfr[nt] = ld_frag32(bp);
      }
#pragma unroll
      for (int mt = 0; mt < 2; ++mt)
#pragma unroll
        for (int nt = 0; nt < 4; ++nt)
          acc[mt][nt] = wmma_bf16(a[mt], bfr[nt], acc[mt][nt]);
    }
  }

#pragma unroll
  for (int mt = 0; mt < 2; ++mt) {
#pragma unroll
    for (int nt = 0; nt < 4; ++nt) {
      const int gn = n0 + nt * 16 + r;
      const float bv = bias[gn];
#pragma unroll
      for (int i = 0; i < 8; ++i) {
        const int gm = m0 + mt * 16 + i + half * 8;
        Y[(size_t)gm * D_ + gn] = acc[mt][nt][i] + bv;
      }
    }
  }
}

// ---------------------------------------------------------------------------
// launch
// ---------------------------------------------------------------------------
extern "C" void kernel_launch(void* const* d_in, const int* in_sizes, int n_in,
                              void* d_out, int out_size, void* d_ws, size_t ws_size,
                              hipStream_t stream) {
  const float* q  = (const float*)d_in[0];
  const float* k  = (const float*)d_in[1];
  const float* v  = (const float*)d_in[2];
  // d_in[3] = causal mask: implemented analytically, unused
  const float* Wq = (const float*)d_in[4];
  const float* bq = (const float*)d_in[5];
  const float* Wk = (const float*)d_in[6];
  const float* bk = (const float*)d_in[7];
  const float* Wv = (const float*)d_in[8];
  const float* bv = (const float*)d_in[9];
  const float* Wo = (const float*)d_in[10];
  const float* bo = (const float*)d_in[11];
  float* out = (float*)d_out;

  char* ws = (char*)d_ws;
  const size_t MB = 1024 * 1024;
  bf16_t* Wpack = (bf16_t*)(ws);              //  8 MB: 4 packed weight mats
  bf16_t* Qb    = (bf16_t*)(ws + 8  * MB);    // 16 MB: [B,H,S,HD]
  bf16_t* Kb    = (bf16_t*)(ws + 24 * MB);    // 16 MB: [B,H,S,HD]
  bf16_t* Vt    = (bf16_t*)(ws + 40 * MB);    // 16 MB: [B,H,HD,S]
  bf16_t* Ob    = (bf16_t*)(ws + 56 * MB);    // 16 MB: [BS,D]

  const size_t WSZ = (size_t)D_ * D_;

  pack_weights<<<1024, 256, 0, stream>>>(Wq, Wk, Wv, Wo, Wpack);
  proj_gemm<0><<<dim3(BS_ / 128, D_ / 128), 256, 0, stream>>>(q, Wpack + 0 * WSZ, bq, Qb);
  proj_gemm<0><<<dim3(BS_ / 128, D_ / 128), 256, 0, stream>>>(k, Wpack + 1 * WSZ, bk, Kb);
  proj_gemm<1><<<dim3(BS_ / 128, D_ / 128), 256, 0, stream>>>(v, Wpack + 2 * WSZ, bv, Vt);
  flash_attn<<<dim3(S_ / 128, B_ * H_), 256, 0, stream>>>(Qb, Kb, Vt, Ob);
  out_gemm<<<dim3(BS_ / 128, D_ / 128), 256, 0, stream>>>(Ob, Wpack + 3 * WSZ, bo, out);
}
